// VectorQuantize_8315056685084
// MI455X (gfx1250) — matrix operand order repaired and checked by reading, not compiled, not run on hardware
//
#include <hip/hip_runtime.h>

typedef unsigned int u32;
typedef __bf16 bf16_t;
typedef bf16_t v16bf __attribute__((ext_vector_type(16)));
typedef float  v8f   __attribute__((ext_vector_type(8)));
typedef int    v4i   __attribute__((ext_vector_type(4)));
typedef __attribute__((address_space(1))) v4i* gv4i_p;   // global int4*
typedef __attribute__((address_space(3))) v4i* lv4i_p;   // LDS int4*

#define NROWS   131072
#define DIMK    512
#define NEMB    1024
#define DECAY   0.8f
#define EPSF    1e-05f

// output layout (flat float elements, reference return order)
#define OFF_Q       0                       // quantize_st: N*DIM
#define OFF_IND     67108864                // embed_ind:   N
#define OFF_LOSS    67239936                // loss:        1
#define OFF_NEWEMB  67239937                // new_embed:   DIM*NEMB
#define OFF_NCS     67764225                // new_cluster_size: NEMB
#define OFF_NEA     67765249                // new_embed_avg: DIM*NEMB

// workspace layout (float element offsets)
#define WS_ENORM    0          // 1024
#define WS_ONEHOT   1024       // 1024
#define WS_TOTAL    2048       // 1
#define WS_LOSS     2049       // 1
#define WS_ESUM     4096       // 524288
#define WS_IND      528384     // 131072 ints
#define WS_SWZ      659456     // 262144 u32 (pre-swizzled bf16 embed fragments)

#if defined(__has_builtin)
#if __has_builtin(__builtin_amdgcn_global_load_async_to_lds_b128) && \
    __has_builtin(__builtin_amdgcn_s_wait_asynccnt)
#define USE_ASYNC_LDS 1
#endif
#endif

__device__ __forceinline__ u32 bf16_of(float f) {
    u32 u = __float_as_uint(f);
    return (u + 0x7FFFu + ((u >> 16) & 1u)) >> 16;   // round-to-nearest-even
}
__device__ __forceinline__ u32 pack2(float a, float b) {
    return bf16_of(a) | (bf16_of(b) << 16);
}

union Frag { v16bf v; u32 u[8]; };

__device__ __forceinline__ void loadFragLDS(Frag& b, const u32* base, int f, int lane) {
    const uint4* p = (const uint4*)(base + (size_t)(f * 32 + lane) * 8);
    uint4 a0 = p[0];
    uint4 a1 = p[1];
    b.u[0] = a0.x; b.u[1] = a0.y; b.u[2] = a0.z; b.u[3] = a0.w;
    b.u[4] = a1.x; b.u[5] = a1.y; b.u[6] = a1.z; b.u[7] = a1.w;
}

// ---------------- kernel 1: column norms of embed ----------------
__global__ void vq_enorm(const float* __restrict__ embed, float* __restrict__ enorm) {
    int j = blockIdx.x * blockDim.x + threadIdx.x;
    if (j < NEMB) {
        float s = 0.f;
        for (int k = 0; k < DIMK; ++k) {
            float e = embed[(size_t)k * NEMB + j];
            s += e * e;
        }
        enorm[j] = s;
    }
}

// ---------------- kernel 2: pre-swizzle embed into bf16 B-fragment layout ----------
// swz dword index d = ((ct*16 + f)*32 + lane)*8 + v
// lane = hh*16 + nl; value packs K = 32f + 8hh + 2(v&3) + 16(v>>2) (+1), column n = 16ct + nl.
__global__ __launch_bounds__(256) void vq_swizzle(const float* __restrict__ embed,
                                                  u32* __restrict__ swz) {
    int d   = blockIdx.x * 256 + threadIdx.x;   // 0..262143
    int ct  = d >> 12;
    int rem = d & 4095;
    int f   = rem >> 8;
    int l   = (rem >> 3) & 31;
    int v   = rem & 7;
    int hh  = l >> 4;
    int nl  = l & 15;
    int k0  = f * 32 + hh * 8 + (v & 3) * 2 + (v >> 2) * 16;
    int n   = ct * 16 + nl;
    swz[d] = pack2(embed[(size_t)k0 * NEMB + n], embed[(size_t)(k0 + 1) * NEMB + n]);
}

// ---------------- kernel 3: WMMA distance GEMM + argmin ----------------
// 8 waves per WG, 16 rows per wave (128 rows per WG). grid = NROWS/128.
__global__ __launch_bounds__(256) void vq_argmin(
    const float* __restrict__ x, const u32* __restrict__ swz,
    const float* __restrict__ enorm,
    int* __restrict__ ind_ws, float* __restrict__ ind_out)
{
    __shared__ u32 ldsB[2][4096];   // double-buffered 16KB B tiles

    const int tid  = threadIdx.x;
    const int wave = tid >> 5;
    const int lane = tid & 31;
    const int h    = lane >> 4;     // lane half
    const int mr   = lane & 15;     // A: row-in-tile; B/C: col-in-tile
    const int rowBase = blockIdx.x * 128 + wave * 16;

    // ---- load A fragments (16 rows x K=512, bf16) into registers, ISA layout ----
    Frag afrag[16];
    const float* xrow = x + (size_t)(rowBase + mr) * DIMK;
    #pragma unroll
    for (int f = 0; f < 16; ++f) {
        const int k0 = f * 32 + h * 8;        // VGPR 0..3: K = k0..k0+7
        const int k1 = k0 + 16;               // VGPR 4..7: K = k1..k1+7
        float4 p0 = *(const float4*)(xrow + k0);
        float4 p1 = *(const float4*)(xrow + k0 + 4);
        float4 p2 = *(const float4*)(xrow + k1);
        float4 p3 = *(const float4*)(xrow + k1 + 4);
        afrag[f].u[0] = pack2(p0.x, p0.y);  afrag[f].u[1] = pack2(p0.z, p0.w);
        afrag[f].u[2] = pack2(p1.x, p1.y);  afrag[f].u[3] = pack2(p1.z, p1.w);
        afrag[f].u[4] = pack2(p2.x, p2.y);  afrag[f].u[5] = pack2(p2.z, p2.w);
        afrag[f].u[6] = pack2(p3.x, p3.y);  afrag[f].u[7] = pack2(p3.z, p3.w);
    }

    // stage a 16KB pre-swizzled tile into LDS (async DMA when available)
    auto stage = [&](int ct, int buf) {
        const uint4* gsrc = (const uint4*)(swz + (size_t)ct * 4096);
        uint4* dst = (uint4*)&ldsB[buf][0];
        #pragma unroll
        for (int it = 0; it < 4; ++it) {
            int i = tid + it * 256;   // 0..1023 uint4
#if defined(USE_ASYNC_LDS)
            __builtin_amdgcn_global_load_async_to_lds_b128(
                (gv4i_p)(gsrc + i), (lv4i_p)(dst + i), 0, 0);
#else
            dst[i] = gsrc[i];
#endif
        }
    };

    float best[8];
    int   bidx[8];
    #pragma unroll
    for (int r = 0; r < 8; ++r) { best[r] = 3.4e38f; bidx[r] = 0; }

    stage(0, 0);
    for (int ct = 0; ct < NEMB / 16; ++ct) {
#if defined(USE_ASYNC_LDS)
        __builtin_amdgcn_s_wait_asynccnt(0);
#endif
        __syncthreads();                 // staging of ct visible; prior compute done
        if (ct + 1 < NEMB / 16) stage(ct + 1, (ct + 1) & 1);

        const u32* base = &ldsB[ct & 1][0];
        Frag bb[2];
        loadFragLDS(bb[0], base, 0, lane);
        v8f acc = {};
        #pragma unroll
        for (int f = 0; f < 16; ++f) {
            if (f < 15) loadFragLDS(bb[(f + 1) & 1], base, f + 1, lane);
            acc = __builtin_amdgcn_wmma_f32_16x16x32_bf16(
                false, afrag[f].v, false, bb[f & 1].v, (short)0, acc, false, false);
        }

        const int   col = ct * 16 + mr;
        const float en  = enorm[col];
        #pragma unroll
        for (int r = 0; r < 8; ++r) {
            float s = en - 2.0f * acc[r];   // ||e||^2 - 2 x.e (||x||^2 invariant)
            if (s < best[r]) { best[r] = s; bidx[r] = col; }
        }
    }

    // reduce (min, argmin) across the 16 lanes of each half (C layout: lane%16 = col)
    #pragma unroll
    for (int off = 8; off >= 1; off >>= 1) {
        #pragma unroll
        for (int r = 0; r < 8; ++r) {
            float ob = __shfl_xor(best[r], off, 32);
            int   oi = __shfl_xor(bidx[r], off, 32);
            if (ob < best[r] || (ob == best[r] && oi < bidx[r])) {
                best[r] = ob; bidx[r] = oi;
            }
        }
    }

    if (mr == 0) {   // lanes 0 and 16: rows r + 8h
        #pragma unroll
        for (int r = 0; r < 8; ++r) {
            int row = rowBase + h * 8 + r;
            ind_ws[row]  = bidx[r];
            ind_out[row] = (float)bidx[r];
        }
    }
}

// ---------------- kernel 4: gather quantize, loss, scatter EMA sums ----------------
__global__ __launch_bounds__(256) void vq_scatter(
    const float* __restrict__ x, const float* __restrict__ target,
    const float* __restrict__ embed, const int* __restrict__ ind,
    float* __restrict__ q_out, float* __restrict__ embed_sum,
    float* __restrict__ onehot, float* __restrict__ loss_acc)
{
    __shared__ float red[256];
    const int row = blockIdx.x;
    const int idx = ind[row];
    float lsum = 0.f;
    for (int d = threadIdx.x; d < DIMK; d += 256) {
        float q = embed[(size_t)d * NEMB + idx];        // embed.T[idx] gather
        float t = target[(size_t)row * DIMK + d];
        q_out[(size_t)row * DIMK + d] = q;              // quantize_st == quantize
        float diff = q - t;
        lsum += diff * diff;
        atomicAdd(&embed_sum[(size_t)d * NEMB + idx], x[(size_t)row * DIMK + d]);
    }
    red[threadIdx.x] = lsum;
    __syncthreads();
    for (int s = 128; s > 0; s >>= 1) {
        if (threadIdx.x < s) red[threadIdx.x] += red[threadIdx.x + s];
        __syncthreads();
    }
    if (threadIdx.x == 0) {
        atomicAdd(loss_acc, red[0]);
        atomicAdd(&onehot[idx], 1.0f);
    }
}

// ---------------- kernel 5: cluster-size EMA + total + loss finalize ----------------
__global__ __launch_bounds__(1024) void vq_cluster(
    const float* __restrict__ cluster_size, const float* __restrict__ onehot,
    const float* __restrict__ loss_acc,
    float* __restrict__ ncs_out, float* __restrict__ loss_out,
    float* __restrict__ total_ws)
{
    __shared__ float red[1024];
    const int j = threadIdx.x;
    float v = cluster_size[j] * DECAY + (1.0f - DECAY) * onehot[j];
    ncs_out[j] = v;
    red[j] = v;
    __syncthreads();
    for (int s = 512; s > 0; s >>= 1) {
        if (j < s) red[j] += red[j + s];
        __syncthreads();
    }
    if (j == 0) {
        total_ws[0] = red[0];
        loss_out[0] = loss_acc[0] * (1.0f / ((float)NROWS * (float)DIMK));
    }
}

// ---------------- kernel 6: embed_avg EMA + smoothed division ----------------
__global__ __launch_bounds__(256) void vq_finalize(
    const float* __restrict__ embed_avg, const float* __restrict__ embed_sum,
    const float* __restrict__ ncs, const float* __restrict__ total_ws,
    float* __restrict__ nea_out, float* __restrict__ nemb_out)
{
    const int e = blockIdx.x * 256 + threadIdx.x;     // DIM*NEMB = 524288
    const float total = total_ws[0];
    const int j = e & (NEMB - 1);
    float nea = embed_avg[e] * DECAY + (1.0f - DECAY) * embed_sum[e];
    nea_out[e] = nea;
    float sm = (ncs[j] + EPSF) / (total + (float)NEMB * EPSF) * total;
    nemb_out[e] = nea / sm;
}

extern "C" void kernel_launch(void* const* d_in, const int* in_sizes, int n_in,
                              void* d_out, int out_size, void* d_ws, size_t ws_size,
                              hipStream_t stream) {
    const float* x            = (const float*)d_in[0];
    const float* target       = (const float*)d_in[1];
    const float* embed        = (const float*)d_in[2];
    const float* cluster_size = (const float*)d_in[3];
    const float* embed_avg    = (const float*)d_in[4];
    float* out = (float*)d_out;
    float* ws  = (float*)d_ws;

    float* enorm     = ws + WS_ENORM;
    float* onehot    = ws + WS_ONEHOT;
    float* total     = ws + WS_TOTAL;
    float* loss_acc  = ws + WS_LOSS;
    float* embed_sum = ws + WS_ESUM;
    int*   ind       = (int*)(ws + WS_IND);
    u32*   swz       = (u32*)(ws + WS_SWZ);

    // zero onehot .. embed_sum (capturable memset node)
    (void)hipMemsetAsync(ws + WS_ONEHOT, 0,
                         (size_t)(WS_ESUM - WS_ONEHOT + DIMK * NEMB) * sizeof(float), stream);

    vq_enorm   <<<NEMB / 256, 256, 0, stream>>>(embed, enorm);
    vq_swizzle <<<(NEMB / 16) * 4096 / 256, 256, 0, stream>>>(embed, swz);
    vq_argmin  <<<NROWS / 128, 256, 0, stream>>>(x, swz, enorm, ind, out + OFF_IND);
    vq_scatter <<<NROWS, 256, 0, stream>>>(x, target, embed, ind,
                                           out + OFF_Q, embed_sum, onehot, loss_acc);
    vq_cluster <<<1, 1024, 0, stream>>>(cluster_size, onehot, loss_acc,
                                        out + OFF_NCS, out + OFF_LOSS, total);
    vq_finalize<<<(DIMK * NEMB) / 256, 256, 0, stream>>>(embed_avg, embed_sum,
                                        out + OFF_NCS, total,
                                        out + OFF_NEA, out + OFF_NEWEMB);
}
